// Imcnn_52347061403772
// MI455X (gfx1250) — matrix operand order repaired and checked by reading, not compiled, not run on hardware
//
#include <hip/hip_runtime.h>

// ---------------------------------------------------------------------------
// IMCNN (FAUST) forward pass for gfx1250 (MI455X): bf16 WMMA + TDM-staged LDS.
// ---------------------------------------------------------------------------

#define NV    6890   // vertices (and dense-head classes)
#define NT    100    // channels per conv layer
#define TP    112    // channel stride (pad 100 -> 112, 7 tiles of 16)
#define NR    5      // radial
#define NA    8      // angular (= number of rotations)
#define NROT  8
#define KS    512    // padded K per angular slice (R*C = 500 -> 512)
#define VTILES 431   // ceil(6890/16)
#define VTG   54     // ceil(431/8) v-tile groups (8 waves per block)
#define TTILES 7     // ceil(100/16)
#define DKP   128    // dense-head K padded (100 -> 128)
#define EPSBN 1e-3f

#if defined(__has_builtin)
#if __has_builtin(__builtin_amdgcn_tensor_load_to_lds)
#define IMCNN_TDM 1
#endif
#endif

typedef __bf16 bf16x8  __attribute__((ext_vector_type(8)));
typedef __bf16 bf16x16 __attribute__((ext_vector_type(16)));
typedef float  f32x8   __attribute__((ext_vector_type(8)));
typedef unsigned short u16x8 __attribute__((ext_vector_type(8)));
typedef unsigned int uint32x4 __attribute__((ext_vector_type(4)));
typedef int int32x4 __attribute__((ext_vector_type(4)));
typedef int int32x8 __attribute__((ext_vector_type(8)));

__device__ __forceinline__ unsigned short f2bf(float f) {
  unsigned int u = __float_as_uint(f);
  u += 0x7FFFu + ((u >> 16) & 1u);          // round-to-nearest-even
  return (unsigned short)(u >> 16);
}

__device__ __forceinline__ bf16x16 cat16(bf16x8 lo, bf16x8 hi) {
  return __builtin_shufflevector(lo, hi, 0,1,2,3,4,5,6,7,8,9,10,11,12,13,14,15);
}

// ---------------------------------------------------------------------------
// Layer 0: C_in = 3, tiny (1.3 GFLOP) -> plain VALU. One block per vertex.
// ---------------------------------------------------------------------------
__global__ __launch_bounds__(128) void layer0_k(
    const float* __restrict__ sig, const int* __restrict__ bidx,
    const float* __restrict__ bw, const float* __restrict__ nmean,
    const float* __restrict__ nvar, const float* __restrict__ W0,
    const float* __restrict__ b0, float* __restrict__ y) {
  const int v = blockIdx.x;
  __shared__ float patch[NR * NA * 3];
  const int tid = threadIdx.x;
  if (tid < NR * NA * 3) {
    const int r = tid / (NA * 3), a = (tid / 3) % NA, c = tid % 3;
    const int base = ((v * NR + r) * NA + a) * 3;
    float acc = 0.f;
#pragma unroll
    for (int i = 0; i < 3; ++i) {
      const int vv = bidx[base + i];
      const float s = (sig[vv * 3 + c] - nmean[c]) * rsqrtf(nvar[c]);
      acc += s * bw[base + i];
    }
    patch[tid] = acc;
  }
  __syncthreads();
  if (tid < NT) {
    const float bias = b0[tid];
#pragma unroll
    for (int n = 0; n < NROT; ++n) {
      float s = 0.f;
      for (int r = 0; r < NR; ++r)
        for (int a = 0; a < NA; ++a) {
          const int ar = (a + n) & 7;
          const float* wp = W0 + ((tid * NR + r) * NA + ar) * 3;
          const float* pp = patch + (r * NA + a) * 3;
          s += pp[0] * wp[0] + pp[1] * wp[1] + pp[2] * wp[2];
        }
      y[((size_t)v * NROT + n) * TP + tid] = fmaxf(s + bias, 0.f);
    }
  }
}

// ---------------------------------------------------------------------------
// Angular max pooling (L2 over channels) + inference BatchNorm.
// ---------------------------------------------------------------------------
__global__ __launch_bounds__(256) void amp_bn_k(
    const float* __restrict__ y, const float* __restrict__ g,
    const float* __restrict__ b, const float* __restrict__ m,
    const float* __restrict__ vv, int layer, float* __restrict__ x) {
  const int v = blockIdx.x;
  const int lane = threadIdx.x & 31, w = threadIdx.x >> 5;
  float s = 0.f;
  for (int t = lane; t < NT; t += 32) {
    const float val = y[((size_t)v * NROT + w) * TP + t];
    s += val * val;
  }
#pragma unroll
  for (int off = 16; off > 0; off >>= 1) s += __shfl_down(s, off);
  __shared__ float norms[NROT];
  __shared__ int best;
  if (lane == 0) norms[w] = s;
  __syncthreads();
  if (threadIdx.x == 0) {
    int bi = 0; float bm = norms[0];
    for (int n = 1; n < NROT; ++n)
      if (norms[n] > bm) { bm = norms[n]; bi = n; }   // first-max like argmax
    best = bi;
  }
  __syncthreads();
  const int nb = best;
  if (threadIdx.x < NT) {
    const int t = threadIdx.x;
    const float val = y[((size_t)v * NROT + nb) * TP + t];
    const float gg = g[layer * NT + t], bb = b[layer * NT + t];
    const float mm = m[layer * NT + t], vr = vv[layer * NT + t];
    x[(size_t)v * TP + t] = gg * (val - mm) * rsqrtf(vr + EPSBN) + bb;
  }
}

// ---------------------------------------------------------------------------
// Barycentric patch gather -> bf16 A-matrix: pack[v][a][k], k = r*100+c (pad 512)
// ---------------------------------------------------------------------------
__global__ __launch_bounds__(256) void pack_patch_k(
    const float* __restrict__ x, const int* __restrict__ bidx,
    const float* __restrict__ bw, unsigned short* __restrict__ pack) {
  const int v = blockIdx.x;
  for (int e = threadIdx.x; e < NA * KS; e += blockDim.x) {
    const int a = e >> 9, kk = e & 511;
    unsigned short outv = 0;
    if (kk < NR * NT) {
      const int r = kk / NT, c = kk - r * NT;
      const int base = ((v * NR + r) * NA + a) * 3;
      float s = x[(size_t)bidx[base + 0] * TP + c] * bw[base + 0]
              + x[(size_t)bidx[base + 1] * TP + c] * bw[base + 1]
              + x[(size_t)bidx[base + 2] * TP + c] * bw[base + 2];
      outv = f2bf(s);
    }
    pack[(size_t)v * (NA * KS) + e] = outv;
  }
}

// ---------------------------------------------------------------------------
// Conv weights -> WMMA B-fragment tiles: w2f[((a'*16+kc)*7+tt)*512 + lane*16 + e]
// ---------------------------------------------------------------------------
__global__ __launch_bounds__(256) void pack_w_k(
    const float* __restrict__ W, unsigned short* __restrict__ w2f) {
  const int gid = blockIdx.x * blockDim.x + threadIdx.x;
  if (gid >= NA * 16 * TTILES * 512) return;
  const int e = gid & 15, lane = (gid >> 4) & 31;
  const int tile = gid >> 9;
  const int tt = tile % TTILES, kc = (tile / TTILES) % 16, a = tile / (TTILES * 16);
  const int k = kc * 32 + (lane >> 4) * 16 + e;
  const int t = tt * 16 + (lane & 15);
  float val = 0.f;
  if (k < NR * NT && t < NT) {
    const int r = k / NT, c = k - r * NT;
    val = W[((t * NR + r) * NA + a) * NT + c];
  }
  w2f[gid] = f2bf(val);
}

// ---------------------------------------------------------------------------
// Multi-rotation WMMA conv. Block = 8 waves = 8 v-tiles sharing one t-tile.
// Weight fragments for this t-tile are staged in LDS in 32KB chunks via the
// Tensor Data Mover (or a cooperative copy fallback), then each wave runs
// 8 accumulators x 8 A-slices against LDS-resident B fragments.
// ---------------------------------------------------------------------------
__global__ __launch_bounds__(256) void conv_wmma_k(
    const unsigned short* __restrict__ pack, const unsigned short* __restrict__ w2f,
    const float* __restrict__ bias, float* __restrict__ y) {
  __shared__ unsigned short smem[NA * 4 * 512];   // 32 KB: 8 a' x 4 kc fragment tiles

  const int lane = threadIdx.x & 31, wave = threadIdx.x >> 5;
  const int vtg = blockIdx.x / TTILES, tt = blockIdx.x - vtg * TTILES;
  const int vt = vtg * 8 + wave;                  // may exceed VTILES-1 (masked stores)
  const int m = lane & 15, kh = lane >> 4;

  int vA = vt * 16 + m; if (vA >= NV) vA = NV - 1;
  const unsigned short* arow = pack + (size_t)vA * (NA * KS);

  f32x8 acc[NROT];
#pragma unroll
  for (int n = 0; n < NROT; ++n) acc[n] = (f32x8){0.f,0.f,0.f,0.f,0.f,0.f,0.f,0.f};

  for (int cg = 0; cg < 4; ++cg) {
    __syncthreads();                              // previous chunk fully consumed
#ifdef IMCNN_TDM
    if (wave == 0) {
      // D# describing a 3D tile: x = 1KB fragment row (128 x 8B),
      // y = 4 kc tiles @ 7KB stride, z = 8 a' slices @ 112KB stride.
      const unsigned long long ga =
          (unsigned long long)(const void*)(w2f + ((size_t)(cg * 28 + tt) << 9));
      const unsigned ldsoff = (unsigned)(unsigned long long)(const void*)smem;
      uint32x4 g0;
      g0[0] = 1u;                                 // count=1 valid user descriptor
      g0[1] = ldsoff;                             // lds_addr (bytes)
      g0[2] = (unsigned)ga;                       // global_addr[31:0]
      g0[3] = (unsigned)(ga >> 32) | (2u << 30);  // global_addr[56:32] | type=2
      int32x8 g1;
      g1[0] = (int)(3u << 16);                    // data_size = 8B
      g1[1] = (int)(128u << 16);                  // tensor_dim0 = 128 (8B units)
      g1[2] = (int)(4u << 16);                    // tensor_dim1 = 4
      g1[3] = (int)(128u << 16);                  // tile_dim0 = 128
      g1[4] = (int)(4u | (8u << 16));             // tile_dim1 = 4, tile_dim2 = 8
      g1[5] = 896;                                // tensor_dim0_stride = 896 (7KB)
      g1[6] = (int)(14336u << 16);                // tensor_dim1_stride = 14336 (112KB)
      g1[7] = 0;
      int32x4 g2; g2[0] = 8; g2[1] = 0; g2[2] = 0; g2[3] = 0;   // tensor_dim2 = 8
      int32x4 g3; g3[0] = 0; g3[1] = 0; g3[2] = 0; g3[3] = 0;
#if __clang_major__ >= 23
      int32x8 g4; g4[0]=0; g4[1]=0; g4[2]=0; g4[3]=0; g4[4]=0; g4[5]=0; g4[6]=0; g4[7]=0;
      __builtin_amdgcn_tensor_load_to_lds(g0, g1, g2, g3, g4, 0);
#else
      __builtin_amdgcn_tensor_load_to_lds(g0, g1, g2, g3, 0);
#endif
      __builtin_amdgcn_s_wait_tensorcnt(0);
    }
#else
    for (int idx = threadIdx.x; idx < 2048; idx += 256) {   // 2048 x 16B = 32KB
      const int tile = idx >> 6, within = idx & 63;         // tile = a*4 + kc2
      const int a = tile >> 2, kc2 = tile & 3;
      const unsigned short* src =
          w2f + (((size_t)((a * 16 + cg * 4 + kc2) * TTILES + tt)) << 9) + within * 8;
      *(u16x8*)(smem + (tile << 9) + within * 8) = *(const u16x8*)src;
    }
#endif
    __syncthreads();                              // chunk visible to all waves

    for (int kc2 = 0; kc2 < 4; ++kc2) {
      const int kc = cg * 4 + kc2;
      bf16x16 bfr[NA];
#pragma unroll
      for (int ap = 0; ap < NA; ++ap) {
        const unsigned short* bp = smem + (((ap << 2) | kc2) << 9) + lane * 16;
        bfr[ap] = cat16(*(const bf16x8*)bp, *(const bf16x8*)(bp + 8));
      }
#pragma unroll
      for (int a = 0; a < NA; ++a) {
        const unsigned short* apt = arow + a * KS + kc * 32 + kh * 16;
        const bf16x16 af = cat16(*(const bf16x8*)apt, *(const bf16x8*)(apt + 8));
#pragma unroll
        for (int n = 0; n < NROT; ++n) {
          acc[n] = __builtin_amdgcn_wmma_f32_16x16x32_bf16(
              false, af, false, bfr[(a + n) & 7], (short)0, acc[n], false, false);
        }
      }
    }
  }

  const int t = tt * 16 + m;
  const float bv = (t < NT) ? bias[t] : 0.f;
#pragma unroll
  for (int n = 0; n < NROT; ++n) {
#pragma unroll
    for (int i = 0; i < 8; ++i) {
      const int v = vt * 16 + kh * 8 + i;
      if (v < NV)
        y[((size_t)v * NROT + n) * TP + t] = fmaxf(acc[n][i] + bv, 0.f);
    }
  }
}

// ---------------------------------------------------------------------------
// Dense head packing + WMMA GEMM: (V,100) @ (100,V), K padded to 128.
// ---------------------------------------------------------------------------
__global__ __launch_bounds__(256) void pack_xbf_k(
    const float* __restrict__ x, unsigned short* __restrict__ xbf) {
  const int gid = blockIdx.x * blockDim.x + threadIdx.x;
  if (gid >= NV * DKP) return;
  const int v = gid >> 7, k = gid & 127;
  xbf[gid] = (k < NT) ? f2bf(x[(size_t)v * TP + k]) : (unsigned short)0;
}

__global__ __launch_bounds__(256) void pack_wd_k(
    const float* __restrict__ dw, unsigned short* __restrict__ wdf) {
  const int gid = blockIdx.x * blockDim.x + threadIdx.x;
  if (gid >= 4 * VTILES * 512) return;
  const int e = gid & 15, lane = (gid >> 4) & 31;
  const int tile = gid >> 9;
  const int nt = tile % VTILES, kc = tile / VTILES;
  const int k = kc * 32 + (lane >> 4) * 16 + e;
  const int n = nt * 16 + (lane & 15);
  float val = 0.f;
  if (k < NT && n < NV) val = dw[(size_t)k * NV + n];
  wdf[gid] = f2bf(val);
}

__global__ __launch_bounds__(256) void dense_wmma_k(
    const unsigned short* __restrict__ xbf, const unsigned short* __restrict__ wdf,
    const float* __restrict__ db, float* __restrict__ out) {
  const int lane = threadIdx.x & 31, wave = threadIdx.x >> 5;
  const int unit = blockIdx.x * 8 + wave;
  if (unit >= VTILES * VTILES) return;
  const int vt = unit / VTILES, nt = unit - vt * VTILES;
  const int m = lane & 15, kh = lane >> 4;
  int vA = vt * 16 + m; if (vA >= NV) vA = NV - 1;
  const unsigned short* ar = xbf + (size_t)vA * DKP;

  f32x8 acc = (f32x8){0.f,0.f,0.f,0.f,0.f,0.f,0.f,0.f};
#pragma unroll
  for (int kc = 0; kc < 4; ++kc) {
    const unsigned short* ap = ar + kc * 32 + kh * 16;
    const bf16x16 af = cat16(*(const bf16x8*)ap, *(const bf16x8*)(ap + 8));
    const unsigned short* bp = wdf + (((size_t)kc * VTILES + nt) << 9) + lane * 16;
    const bf16x16 bf = cat16(*(const bf16x8*)bp, *(const bf16x8*)(bp + 8));
    acc = __builtin_amdgcn_wmma_f32_16x16x32_bf16(
        false, af, false, bf, (short)0, acc, false, false);
  }
  const int n = nt * 16 + m;
  if (n < NV) {
    const float bv = db[n];
#pragma unroll
    for (int i = 0; i < 8; ++i) {
      const int v = vt * 16 + kh * 8 + i;
      if (v < NV) out[(size_t)v * NV + n] = acc[i] + bv;
    }
  }
}

// ---------------------------------------------------------------------------
// Host orchestration (all launches on `stream`, graph-capture safe).
// ---------------------------------------------------------------------------
static inline size_t align256(size_t b) { return (b + 255) & ~(size_t)255; }

extern "C" void kernel_launch(void* const* d_in, const int* in_sizes, int n_in,
                              void* d_out, int out_size, void* d_ws, size_t ws_size,
                              hipStream_t stream) {
  (void)in_sizes; (void)n_in; (void)out_size; (void)ws_size;
  const float* sig   = (const float*)d_in[0];
  const int*   bidx  = (const int*)d_in[1];
  const float* bw    = (const float*)d_in[2];
  const float* nmean = (const float*)d_in[3];
  const float* nvar  = (const float*)d_in[4];
  const float* w0 = (const float*)d_in[5];
  const float* b0 = (const float*)d_in[6];
  const float* w1 = (const float*)d_in[7];
  const float* b1 = (const float*)d_in[8];
  const float* w2 = (const float*)d_in[9];
  const float* b2 = (const float*)d_in[10];
  const float* bng = (const float*)d_in[11];
  const float* bnb = (const float*)d_in[12];
  const float* bnm = (const float*)d_in[13];
  const float* bnv = (const float*)d_in[14];
  const float* dw = (const float*)d_in[15];
  const float* db = (const float*)d_in[16];
  float* out = (float*)d_out;

  char* ws = (char*)d_ws;
  size_t off = 0;
  float* y = (float*)(ws + off);            off += align256((size_t)NV * NROT * TP * 4);
  float* x = (float*)(ws + off);            off += align256((size_t)NV * TP * 4);
  unsigned short* pack = (unsigned short*)(ws + off); off += align256((size_t)NV * NA * KS * 2);
  unsigned short* w2f  = (unsigned short*)(ws + off); off += align256((size_t)NA * 16 * TTILES * 512 * 2);
  unsigned short* xbf  = (unsigned short*)(ws + off); off += align256((size_t)NV * DKP * 2);
  unsigned short* wdf  = (unsigned short*)(ws + off); off += align256((size_t)4 * VTILES * 512 * 2);

  const int convBlocks  = VTG * TTILES;                    // 378
  const int denseBlocks = (VTILES * VTILES + 7) / 8;       // 23221
  const int packWBlocks = (NA * 16 * TTILES * 512) / 256;  // 1792
  const int xbfBlocks   = (NV * DKP + 255) / 256;          // 3445
  const int wdBlocks    = (4 * VTILES * 512 + 255) / 256;  // 3448

  layer0_k<<<NV, 128, 0, stream>>>(sig, bidx, bw, nmean, nvar, w0, b0, y);
  amp_bn_k<<<NV, 256, 0, stream>>>(y, bng, bnb, bnm, bnv, 0, x);

  for (int l = 0; l < 2; ++l) {
    pack_patch_k<<<NV, 256, 0, stream>>>(x, bidx, bw, pack);
    pack_w_k<<<packWBlocks, 256, 0, stream>>>(l == 0 ? w1 : w2, w2f);
    conv_wmma_k<<<convBlocks, 256, 0, stream>>>(pack, w2f, l == 0 ? b1 : b2, y);
    amp_bn_k<<<NV, 256, 0, stream>>>(y, bng, bnb, bnm, bnv, l + 1, x);
  }

  pack_xbf_k<<<xbfBlocks, 256, 0, stream>>>(x, xbf);
  pack_wd_k<<<wdBlocks, 256, 0, stream>>>(dw, wdf);
  dense_wmma_k<<<denseBlocks, 256, 0, stream>>>(xbf, wdf, db, out);
}
